// HyenaOperatorV2_53377853555190
// MI455X (gfx1250) — compile-verified, matrix operand order checked by reading, not run on hardware
//
#include <hip/hip_runtime.h>
#include <hip/hip_bf16.h>
#include <math.h>

// Problem constants (match reference)
#define DIMC 256
#define SEQ  8192
#define NB   4

typedef __attribute__((ext_vector_type(16))) __bf16 v16bf;
typedef __attribute__((ext_vector_type(4)))  __bf16 v4bf;
typedef __attribute__((ext_vector_type(8)))  float  v8f;
typedef __attribute__((ext_vector_type(4)))  unsigned int u32x4;
typedef __attribute__((ext_vector_type(8)))  int i32x8;
typedef __attribute__((ext_vector_type(4)))  int i32x4;

#if defined(__AMDGCN__) && __has_builtin(__builtin_amdgcn_tensor_load_to_lds) && \
    __has_builtin(__builtin_amdgcn_s_wait_tensorcnt)
#define USE_TDM 1
#else
#define USE_TDM 0
#endif

// ---------- helpers ----------

__device__ __forceinline__ __bf16 f2bf(float f) {
  union { float f; unsigned u; } a; a.f = f;
  unsigned r = a.u + 0x7FFFu + ((a.u >> 16) & 1u);   // round-to-nearest-even
  unsigned short h = (unsigned short)(r >> 16);
  __bf16 b;
  __builtin_memcpy(&b, &h, sizeof(h));
  return b;
}

__device__ __forceinline__ float gelu_exact(float v) {
  return 0.5f * v * (1.0f + erff(v * 0.70710678118654752f));
}

#if USE_TDM
// Issue one TDM descriptor: 2-D tile of f32 (tile_x consecutive elems per row,
// tile_y rows at row_stride elems) from global -> LDS. Optional LDS padding:
// pad_enable inserts `pad_amount+1`-encoded DWORDs every 2<<pad_interval DWORDs.
// D# layout per CDNA5 ISA ch.8.
__device__ __forceinline__ void tdm_load_tile_f32(unsigned lds_off,
                                                  const float* gsrc,
                                                  int tile_x, int tile_y,
                                                  int row_stride,
                                                  int pad_en, int pad_interval) {
  unsigned long long ga = (unsigned long long)(size_t)gsrc;
  u32x4 g0;
  g0[0] = 1u;                                          // count=1 (valid), user mode
  g0[1] = lds_off;                                     // LDS byte address
  g0[2] = (unsigned)(ga & 0xFFFFFFFFu);                // global_addr[31:0]
  g0[3] = ((unsigned)(ga >> 32) & 0x01FFFFFFu)         // global_addr[56:32]
        | (2u << 30);                                  // type=2 ("image")
  i32x8 g1;
  g1[0] = (int)((2u << 16)                             // data_size=2 (4 bytes)
        | ((unsigned)(pad_en & 1) << 20)               // pad_enable
        | ((unsigned)(pad_interval & 7) << 22));       // pad_interval; pad_amount=0 (1 DW)
  g1[1] = (int)(((unsigned)tile_x & 0xFFFFu) << 16);   // tensor_dim0[15:0]
  g1[2] = (int)((((unsigned)tile_x >> 16) & 0xFFFFu)   // tensor_dim0[31:16]
        | (((unsigned)tile_y & 0xFFFFu) << 16));       // tensor_dim1[15:0]
  g1[3] = (int)((((unsigned)tile_y >> 16) & 0xFFFFu)   // tensor_dim1[31:16]
        | (((unsigned)tile_x & 0xFFFFu) << 16));       // tile_dim0
  g1[4] = (int)((unsigned)tile_y & 0xFFFFu);           // tile_dim1; tile_dim2=0
  g1[5] = row_stride;                                  // tensor_dim0_stride[31:0]
  g1[6] = 0;
  g1[7] = 0;
  i32x4 z4 = {0, 0, 0, 0};
#if __clang_major__ >= 23
  i32x8 z8 = {0, 0, 0, 0, 0, 0, 0, 0};
  __builtin_amdgcn_tensor_load_to_lds(g0, g1, z4, z4, z8, 0);
#else
  __builtin_amdgcn_tensor_load_to_lds(g0, g1, z4, z4, 0);
#endif
}
#endif  // USE_TDM

// ---------- kernel 0: bulk f32 -> bf16 conversion (memory bound prep pass) ----------
__global__ __launch_bounds__(256)
void cvt_f32_bf16_kernel(const float* __restrict__ src, __bf16* __restrict__ dst,
                         int n4) {
  int i = blockIdx.x * 256 + threadIdx.x;
  const int stride = gridDim.x * 256;
  for (; i < n4; i += stride) {
    float4 f = ((const float4*)src)[i];
    v4bf o;
    o[0] = f2bf(f.x); o[1] = f2bf(f.y); o[2] = f2bf(f.z); o[3] = f2bf(f.w);
    ((v4bf*)dst)[i] = o;
  }
}

// ---------- kernel 1: filter projection  hT[c][t] = sum_l kp[c][l]*Wk[t][l] + bk[t] ----------
// GEMM M=256 (c), N=8192 (t), K=8192 (l), bf16 inputs. Each wave computes a 64x16
// block: four M-tiles share one B (Wk) fragment -> 5 fragment loads per 4 WMMAs.
// k-loop is manually double-buffered so loads for k+32 are in flight across the
// WMMA issue of step k.
__global__ __launch_bounds__(256)
void hproj_wmma_kernel(const __bf16* __restrict__ kp,   // [DIMC][SEQ] bf16
                       const __bf16* __restrict__ Wk,   // [SEQ][SEQ]  bf16
                       const float*  __restrict__ bk,   // [SEQ]
                       float* __restrict__ hT)          // [DIMC][SEQ]
{
  const int lane = threadIdx.x & 31;
  const int wave = threadIdx.x >> 5;
  const int n0 = (blockIdx.x * 2 + (wave >> 2)) * 16;  // t-tile (512 total)
  const int m0 = (wave & 3) * 64;                      // four c-tiles per wave
  const int ksel = (lane < 16) ? 0 : 16;               // K half select

  const __bf16* arow = kp + (size_t)(m0 + (lane & 15)) * SEQ + ksel;
  const __bf16* bp   = Wk + (size_t)(n0 + (lane & 15)) * SEQ + ksel;

  const v8f vzero = {};
  v8f acc[4] = {vzero, vzero, vzero, vzero};

  v16bf bcur = *(const v16bf*)bp;
  v16bf acur[4];
#pragma unroll
  for (int j = 0; j < 4; ++j)
    acur[j] = *(const v16bf*)(arow + (size_t)j * 16 * SEQ);

  for (int k0 = 32; k0 < SEQ; k0 += 32) {
    v16bf bnext = *(const v16bf*)(bp + k0);
    v16bf anext[4];
#pragma unroll
    for (int j = 0; j < 4; ++j)
      anext[j] = *(const v16bf*)(arow + (size_t)j * 16 * SEQ + k0);
    if (k0 + 512 < SEQ) {
      __builtin_prefetch(bp + k0 + 512, 0, 1);    // global_prefetch_b8
      __builtin_prefetch(arow + k0 + 512, 0, 1);
    }
#pragma unroll
    for (int j = 0; j < 4; ++j)
      acc[j] = __builtin_amdgcn_wmma_f32_16x16x32_bf16(false, acur[j], false, bcur,
                                                       (short)0, acc[j], false, false);
    bcur = bnext;
#pragma unroll
    for (int j = 0; j < 4; ++j) acur[j] = anext[j];
  }
#pragma unroll
  for (int j = 0; j < 4; ++j)
    acc[j] = __builtin_amdgcn_wmma_f32_16x16x32_bf16(false, acur[j], false, bcur,
                                                     (short)0, acc[j], false, false);

  // D layout: lane -> n (mod 16), vgpr r -> m (+8 for upper half-lanes)
  const int n  = lane & 15;
  const int mb = (lane < 16) ? 0 : 8;
  const float bias = bk[n0 + n];
#pragma unroll
  for (int j = 0; j < 4; ++j) {
#pragma unroll
    for (int r = 0; r < 8; ++r) {
      hT[(size_t)(m0 + j * 16 + mb + r) * SEQ + (n0 + n)] = acc[j][r] + bias;
    }
  }
}

// ---------- kernel 2: v = gelu(x@Wv.T+bv), z = x@Wz.T+bz ----------
// GEMM M=NB*SEQ, N=DIMC, K=DIMC, bf16 inputs. One wave per 16-row m-tile; A held in
// registers (64 VGPRs) and reused across all 16 n-tiles and both weight matrices.
__global__ __launch_bounds__(256)
void vz_wmma_kernel(const __bf16* __restrict__ x,    // [NB*SEQ][DIMC] bf16
                    const __bf16* __restrict__ Wv, const float* __restrict__ bv,
                    const __bf16* __restrict__ Wz, const float* __restrict__ bz,
                    float* __restrict__ vT,           // [NB][DIMC][SEQ]
                    float* __restrict__ zf)           // [NB*SEQ][DIMC]
{
  const int lane = threadIdx.x & 31;
  const int wave = threadIdx.x >> 5;
  const int mt = blockIdx.x * 8 + wave;   // 0..2047
  const int m0 = mt * 16;

  const int ksel = (lane < 16) ? 0 : 16;
  const __bf16* ap = x + (size_t)(m0 + (lane & 15)) * DIMC + ksel;

  v16bf afrag[8];
#pragma unroll
  for (int kk = 0; kk < 8; ++kk) afrag[kk] = *(const v16bf*)(ap + kk * 32);

  const int n_lane = lane & 15;
  const int mb     = (lane < 16) ? 0 : 8;
  const int bglob  = m0 / SEQ;              // tiles never straddle batch boundary
  const int tbase  = (m0 % SEQ) + mb;

  for (int nt = 0; nt < DIMC / 16; ++nt) {
    const int n0 = nt * 16;
    const __bf16* bvp = Wv + (size_t)(n0 + (lane & 15)) * DIMC + ksel;
    const __bf16* bzp = Wz + (size_t)(n0 + (lane & 15)) * DIMC + ksel;
    v8f accv = {}, accz = {};
#pragma unroll
    for (int kk = 0; kk < 8; ++kk) {
      v16bf bfv = *(const v16bf*)(bvp + kk * 32);
      v16bf bfz = *(const v16bf*)(bzp + kk * 32);
      accv = __builtin_amdgcn_wmma_f32_16x16x32_bf16(false, afrag[kk], false, bfv,
                                                     (short)0, accv, false, false);
      accz = __builtin_amdgcn_wmma_f32_16x16x32_bf16(false, afrag[kk], false, bfz,
                                                     (short)0, accz, false, false);
    }
    const int c = n0 + n_lane;
    const float biasv = bv[c];
    const float biasz = bz[c];
#pragma unroll
    for (int r = 0; r < 8; ++r) {
      float vv = gelu_exact(accv[r] + biasv);
      vT[((size_t)bglob * DIMC + c) * SEQ + (tbase + r)] = vv;   // lane-contig in t
      zf[(size_t)(m0 + mb + r) * DIMC + c] = accz[r] + biasz;    // coalesced in c
    }
  }
}

// ---------- kernel 3: causal long conv  y[b][c][t] = sum_{s<=t} h[t-s][c]*v[b][s][c] ----------
// One workgroup per (b,c): both length-8192 rows live in 64KB of the 320KB WGP LDS,
// filled by the Tensor Data Mover (tensor_load_to_lds + s_wait_tensorcnt) when
// available. Rolling register window of h: 8 FMAs per (1 LDS v read + 1 LDS h read).
__global__ __launch_bounds__(256)
void longconv_kernel(const float* __restrict__ hT,   // [DIMC][SEQ]
                     const float* __restrict__ vT,   // [NB][DIMC][SEQ]
                     float* __restrict__ yT)         // [NB][DIMC][SEQ]
{
  __shared__ float sh[SEQ];   // 32 KB
  __shared__ float sv[SEQ];   // 32 KB
  const int bc = blockIdx.x;            // b*DIMC + c
  const int c  = bc & (DIMC - 1);
  const float* hrow = hT + (size_t)c * SEQ;
  const float* vrow = vT + (size_t)bc * SEQ;

#if USE_TDM
  if (threadIdx.x < 32) {               // wave 0 issues both tensor DMAs
    tdm_load_tile_f32((unsigned)(size_t)(void*)sh, hrow, SEQ, 1, SEQ, 0, 0);
    tdm_load_tile_f32((unsigned)(size_t)(void*)sv, vrow, SEQ, 1, SEQ, 0, 0);
    __builtin_amdgcn_s_wait_tensorcnt(0);
  }
  __syncthreads();
#else
  for (int i = threadIdx.x; i < SEQ; i += 256) {
    sh[i] = hrow[i];
    sv[i] = vrow[i];
  }
  __syncthreads();
#endif

  float* yrow = yT + (size_t)bc * SEQ;
#pragma unroll
  for (int g = 0; g < SEQ / (256 * 8); ++g) {          // 4 groups of 8 t per thread
    const int tg = (g * 256 + threadIdx.x) * 8;
    float acc[8] = {0.f, 0.f, 0.f, 0.f, 0.f, 0.f, 0.f, 0.f};
    float hw[8];
#pragma unroll
    for (int r = 0; r < 8; ++r) hw[r] = sh[tg + r];
    // main region: s <= tg  =>  all 8 lags in-range
#pragma unroll 8
    for (int s = 0; s <= tg; ++s) {
      const float vs = sv[s];
#pragma unroll
      for (int r = 0; r < 8; ++r) acc[r] = fmaf(vs, hw[r], acc[r]);
#pragma unroll
      for (int r = 7; r >= 1; --r) hw[r] = hw[r - 1];   // shift window down
      hw[0] = (s < tg) ? sh[tg - s - 1] : 0.f;
    }
    // tail: s in (tg, tg+7], partial lags
    for (int s = tg + 1; s < tg + 8; ++s) {
      const float vs = sv[s];
#pragma unroll
      for (int r = 0; r < 8; ++r) {
        const int d = tg + r - s;
        if (d >= 0) acc[r] = fmaf(vs, sh[d], acc[r]);
      }
    }
#pragma unroll
    for (int r = 0; r < 8; ++r) yrow[tg + r] = acc[r];
  }
}

// ---------- kernel 4: out = shortconv(x) + y * z  (TDM-transposed yT tile) ----------
// The 64t x 64c yT tile (row stride SEQ) is fetched by the TDM with LDS padding
// (pad every 64 DWORDs by 1 DWORD) to produce the bank-conflict-free [64][65]
// layout directly; fallback is a cooperative strided load.
__global__ __launch_bounds__(256)
void combine_kernel(const float* __restrict__ x,       // [NB][SEQ][DIMC]
                    const float* __restrict__ conv_w,  // [DIMC][1][3]
                    const float* __restrict__ conv_b,  // [DIMC]
                    const float* __restrict__ zf,      // [NB][SEQ][DIMC]
                    const float* __restrict__ yT,      // [NB][DIMC][SEQ]
                    float* __restrict__ out)           // [NB][SEQ][DIMC]
{
  __shared__ float ty[64][65];     // [c][t] tile, padded against bank conflicts
  const int b      = blockIdx.y;
  const int tileId = blockIdx.x;                 // 128 t-tiles * 4 c-tiles
  const int c0 = (tileId & 3) * 64;
  const int t0 = (tileId >> 2) * 64;
  const float* ysrc = yT + ((size_t)b * DIMC + c0) * SEQ + t0;

#if USE_TDM
  if (threadIdx.x < 32) {
    // 64x64 f32 tile, row stride SEQ; pad_interval code 5 = 64 DWORDs, 1 DWORD pad.
    tdm_load_tile_f32((unsigned)(size_t)(void*)&ty[0][0], ysrc, 64, 64, SEQ, 1, 5);
    __builtin_amdgcn_s_wait_tensorcnt(0);
  }
  __syncthreads();
#else
  for (int k = threadIdx.x; k < 64 * 64; k += 256) {
    const int i = k >> 6;        // c offset
    const int j = k & 63;        // t offset (coalesced)
    ty[i][j] = ysrc[(size_t)i * SEQ + j];
  }
  __syncthreads();
#endif

  for (int k = threadIdx.x; k < 64 * 64; k += 256) {
    const int tt = k >> 6;
    const int cc = k & 63;       // c fastest -> coalesced out
    const int t = t0 + tt;
    const int c = c0 + cc;
    const size_t base = ((size_t)b * SEQ + t) * DIMC + c;
    const float x0 = x[base];
    const float x1 = (t >= 1) ? x[base - DIMC]     : 0.f;
    const float x2 = (t >= 2) ? x[base - 2 * DIMC] : 0.f;
    const float xs = conv_b[c] + conv_w[c * 3 + 0] * x2
                               + conv_w[c * 3 + 1] * x1
                               + conv_w[c * 3 + 2] * x0;
    out[base] = xs + ty[cc][tt] * zf[base];
  }
}

// ---------- launcher ----------
extern "C" void kernel_launch(void* const* d_in, const int* in_sizes, int n_in,
                              void* d_out, int out_size, void* d_ws, size_t ws_size,
                              hipStream_t stream) {
  (void)in_sizes; (void)n_in; (void)out_size; (void)ws_size;
  const float* x      = (const float*)d_in[0];
  const float* conv_w = (const float*)d_in[1];
  const float* conv_b = (const float*)d_in[2];
  const float* Wv     = (const float*)d_in[3];
  const float* bv     = (const float*)d_in[4];
  const float* Wz     = (const float*)d_in[5];
  const float* bz     = (const float*)d_in[6];
  const float* kp     = (const float*)d_in[7];   // [DIMC][SEQ][1]
  const float* Wk     = (const float*)d_in[8];   // [SEQ][SEQ]
  const float* bk     = (const float*)d_in[9];
  float* out = (float*)d_out;

  char* ws = (char*)d_ws;
  const size_t MB = (size_t)1 << 20;
  float*  hT   = (float*)(ws);                  //   8 MB [DIMC][SEQ] f32
  float*  vT   = (float*)(ws + 8   * MB);       //  32 MB [NB][DIMC][SEQ] f32
  float*  zf   = (float*)(ws + 40  * MB);       //  32 MB [NB][SEQ][DIMC] f32
  float*  yT   = (float*)(ws + 72  * MB);       //  32 MB [NB][DIMC][SEQ] f32
  __bf16* Wkbf = (__bf16*)(ws + 104 * MB);      // 128 MB [SEQ][SEQ] bf16
  __bf16* kpbf = (__bf16*)(ws + 232 * MB);      //   4 MB [DIMC][SEQ] bf16
  __bf16* xbf  = (__bf16*)(ws + 236 * MB);      //  16 MB [NB*SEQ][DIMC] bf16
  __bf16* Wvbf = (__bf16*)(ws + 252 * MB);      // 128 KB
  __bf16* Wzbf = (__bf16*)(ws + 252 * MB + (1 << 17));  // 128 KB

  // Prep: one-shot bf16 shadows of all WMMA operands (memory bound).
  cvt_f32_bf16_kernel<<<dim3(4096), 256, 0, stream>>>(Wk, Wkbf, (SEQ * SEQ) / 4);
  cvt_f32_bf16_kernel<<<dim3(1024), 256, 0, stream>>>(x,  xbf,  (NB * SEQ * DIMC) / 4);
  cvt_f32_bf16_kernel<<<dim3(512),  256, 0, stream>>>(kp, kpbf, (DIMC * SEQ) / 4);
  cvt_f32_bf16_kernel<<<dim3(64),   256, 0, stream>>>(Wv, Wvbf, (DIMC * DIMC) / 4);
  cvt_f32_bf16_kernel<<<dim3(64),   256, 0, stream>>>(Wz, Wzbf, (DIMC * DIMC) / 4);

  hproj_wmma_kernel<<<dim3(256), 256, 0, stream>>>(kpbf, Wkbf, bk, hT);
  vz_wmma_kernel<<<dim3(256), 256, 0, stream>>>(xbf, Wvbf, bv, Wzbf, bz, vT, zf);
  longconv_kernel<<<dim3(NB * DIMC), 256, 0, stream>>>(hT, vT, yT);
  combine_kernel<<<dim3(512, NB), 256, 0, stream>>>(x, conv_w, conv_b, zf, yT, out);
}
